// RopelessMLAEncoderBlock_50251117363437
// MI455X (gfx1250) — matrix operand order repaired and checked by reading, not compiled, not run on hardware
//
#include <hip/hip_runtime.h>
#include <hip/hip_bf16.h>
#include <stdint.h>

// ---------------- problem constants (match reference) ----------------
constexpr int kE = 1024;     // EMB
constexpr int kH = 16;       // HEADS
constexpr int kD = 64;       // HEAD_DIM
constexpr int kL = 256;      // LATENT
constexpr int kF = 4096;     // FFN
constexpr int kB = 2;
constexpr int kN = 2048;
constexpr int kM = kB * kN;  // 4096 token rows

typedef __attribute__((ext_vector_type(16))) __bf16 v16bf;
typedef __attribute__((ext_vector_type(8)))  float  v8f;
typedef unsigned int u32x4 __attribute__((ext_vector_type(4)));
typedef int          i32x4 __attribute__((ext_vector_type(4)));
typedef int          i32x8 __attribute__((ext_vector_type(8)));

union FragB { v16bf v; uint32_t u[8]; };

__device__ __forceinline__ uint16_t f32_to_bf16(float f) {
  uint32_t x = __float_as_uint(f);
  return (uint16_t)((x + 0x7FFFu + ((x >> 16) & 1u)) >> 16);   // RNE
}

__device__ __forceinline__ v8f v8f_zero() {
  v8f z = {0.f, 0.f, 0.f, 0.f, 0.f, 0.f, 0.f, 0.f};
  return z;
}

// ---------------- Tensor Data Mover: 2D bf16 tile -> LDS (ISA 08 §8.3/8.4) ----
// tile_x elems contiguous (dim0), tile_y rows at stride_elems, data_size = 2B.
// pad_bits: D# group1 word0 pad config (pad_enable|interval|amount) so the LDS
// destination keeps our padded row stride.
__device__ __forceinline__ void tdm_load_2d_bf16(uint32_t lds_byte_off,
                                                 const void* gptr,
                                                 int tile_x, int tile_y,
                                                 int stride_elems,
                                                 uint32_t pad_bits) {
  uint64_t ga = (uint64_t)(uintptr_t)gptr;
  u32x4 g0;
  g0.x = 0x1u;                                   // count=1 (valid), user mode
  g0.y = lds_byte_off;                           // D# lds_addr
  g0.z = (uint32_t)ga;                           // global_addr[31:0]
  g0.w = (uint32_t)((ga >> 32) & 0x01FFFFFFu) | (2u << 30);  // addr[56:32], type=2
  i32x8 g1;
  g1[0] = (int)((1u << 16) | pad_bits);          // wg_mask=0, data_size=2B, pad cfg
  g1[1] = (int)(((uint32_t)tile_x & 0xFFFFu) << 16);          // tensor_dim0[15:0]
  g1[2] = (int)((((uint32_t)tile_x) >> 16) |
                (((uint32_t)tile_y & 0xFFFFu) << 16));        // td0 hi | td1 lo
  g1[3] = (int)((((uint32_t)tile_y) >> 16) |
                (((uint32_t)tile_x) << 16));                  // td1 hi | tile_dim0
  g1[4] = (int)((uint32_t)tile_y & 0xFFFFu);                  // tile_dim1, tile_dim2=0
  g1[5] = (int)(uint32_t)stride_elems;                        // tensor_dim0_stride lo
  g1[6] = 0;                                                  // stride0 hi | stride1 lo
  g1[7] = 0;
  i32x4 z4 = {0, 0, 0, 0};
#if defined(__clang_major__) && __clang_major__ >= 23
  i32x8 z8 = {0, 0, 0, 0, 0, 0, 0, 0};
  __builtin_amdgcn_tensor_load_to_lds(g0, g1, z4, z4, z8, 0);
#else
  __builtin_amdgcn_tensor_load_to_lds(g0, g1, z4, z4, 0);
#endif
}

// pad configs: rows of 64B + 32B pad (GEMM As, 96B stride): interval 16DW -> 3, amount 8DW -> 7
constexpr uint32_t kPad64to96   = (1u << 20) | (3u << 22) | (7u << 25);
// rows of 128B + 16B pad (attention tiles, 144B stride): interval 32DW -> 4, amount 4DW -> 3
constexpr uint32_t kPad128to144 = (1u << 20) | (4u << 22) | (3u << 25);

// ---------------- fp32 -> bf16 cast ----------------
__global__ void __launch_bounds__(256)
cast_f32_bf16_kernel(const float* __restrict__ in, uint16_t* __restrict__ out, int n) {
  int base = blockIdx.x * 1024 + threadIdx.x;
#pragma unroll
  for (int k = 0; k < 4; ++k) {
    int i = base + k * 256;
    if (i < n) out[i] = f32_to_bf16(in[i]);
  }
}

// ---------------- generic WMMA GEMM: C = A[M,K] * B[K,N] + bias ----------------
// bf16 in, fp32 accum. 128x128 tile, BK=32, 8 waves; wave computes 32x64 (2x4 frags).
// A tile staged by TDM (per-wave 16-row descriptors, double-buffered, TENSORcnt);
// B tile staged manually transposed into Bs[n][k] so K-pairs are contiguous.
template <bool GELU>
__global__ void __launch_bounds__(256)
gemm_bf16_kernel(const uint16_t* __restrict__ A, const uint16_t* __restrict__ Bm,
                 const float* __restrict__ bias,
                 float* __restrict__ Cf, uint16_t* __restrict__ Cb,
                 int Md, int Nd, int Kd) {
  __shared__ __attribute__((aligned(16))) uint16_t As[2][128][48];  // [m][k], 96B rows
  __shared__ __attribute__((aligned(16))) uint16_t Bs[2][128][48];  // [n][k] transposed

  const int tid  = threadIdx.x;
  const int lane = tid & 31, wave = tid >> 5;
  const int hl   = lane >> 4, l16 = lane & 15;
  const int wr   = wave >> 1, wc = wave & 1;   // 4x2 wave grid
  const int m0   = blockIdx.y * 128, n0 = blockIdx.x * 128;
  const uint32_t asBase = (uint32_t)(uintptr_t)&As[0][0][0];
  constexpr uint32_t kBufBytes = 128 * 48 * 2;

  v8f acc[2][4];
#pragma unroll
  for (int i = 0; i < 2; ++i)
#pragma unroll
    for (int j = 0; j < 4; ++j) acc[i][j] = v8f_zero();

  auto stage = [&](int k0, int buf) {
    // async TDM: this wave DMAs its 16 rows of the A tile into padded LDS
    tdm_load_2d_bf16(asBase + buf * kBufBytes + (uint32_t)wave * 16 * 96,
                     A + (size_t)(m0 + wave * 16) * Kd + k0,
                     32, 16, Kd, kPad64to96);
    // manual transposed staging of B tile 32x128
    for (int i = tid; i < 32 * 16; i += 256) {
      int kk = i >> 4, c = i & 15;
      uint4 d = *(const uint4*)(Bm + (size_t)(k0 + kk) * Nd + n0 + c * 8);
      const uint16_t* ds = (const uint16_t*)&d;
#pragma unroll
      for (int e = 0; e < 8; ++e) Bs[buf][c * 8 + e][kk] = ds[e];
    }
  };

  const int nk = Kd / 32;
  stage(0, 0);
  __builtin_amdgcn_s_wait_tensorcnt(0);
  __syncthreads();

  for (int t = 0; t < nk; ++t) {
    const int cur = t & 1;
    if (t + 1 < nk) {
      stage((t + 1) * 32, cur ^ 1);   // overlap DMA of next tile with compute
      __builtin_prefetch(Bm + (size_t)((t + 1) * 32 + (tid >> 4)) * Nd + n0 + (tid & 15) * 8, 0, 1);
    }

    // A fragments (16x32 bf16 layout: lanes 0-15 K=0..15, lanes 16-31 K=16..31)
    FragB a[2], b[4];
    const int kbA = hl ? 16 : 0;
#pragma unroll
    for (int i = 0; i < 2; ++i) {
      int row = wr * 32 + i * 16 + l16;
#pragma unroll
      for (int v = 0; v < 4; ++v) {
        a[i].u[v]     = *(const uint32_t*)&As[cur][row][kbA + 2 * v];
        a[i].u[v + 4] = *(const uint32_t*)&As[cur][row][8 + kbA + 2 * v];
      }
    }
    // B fragments (32x16: lanes 0-15 K=0..15, lanes 16-31 K=16..31)
    const int kbB = hl ? 16 : 0;
#pragma unroll
    for (int j = 0; j < 4; ++j) {
      int col = wc * 64 + j * 16 + l16;
#pragma unroll
      for (int v = 0; v < 8; ++v)
        b[j].u[v] = *(const uint32_t*)&Bs[cur][col][kbB + 2 * v];
    }
#pragma unroll
    for (int i = 0; i < 2; ++i)
#pragma unroll
      for (int j = 0; j < 4; ++j)
        acc[i][j] = __builtin_amdgcn_wmma_f32_16x16x32_bf16(
            false, a[i].v, false, b[j].v, (short)0, acc[i][j], false, false);

    __builtin_amdgcn_s_wait_tensorcnt(0);  // next tile's DMA done (issued pre-compute)
    __syncthreads();
  }

  // epilogue: C layout VGPR e -> row e (+8 upper half-wave), lane -> col
#pragma unroll
  for (int i = 0; i < 2; ++i)
#pragma unroll
    for (int j = 0; j < 4; ++j)
#pragma unroll
      for (int e = 0; e < 8; ++e) {
        int row = m0 + wr * 32 + i * 16 + e + 8 * hl;
        int col = n0 + wc * 64 + j * 16 + l16;
        float vv = acc[i][j][e] + bias[col];
        if (GELU) vv = 0.5f * vv * (1.f + erff(vv * 0.70710678118654752f));
        if (Cf) Cf[(size_t)row * Nd + col] = vv;
        if (Cb) Cb[(size_t)row * Nd + col] = f32_to_bf16(vv);
      }
}

// ---------------- fused flash attention ----------------
// Q,K,V stored [B,N,H,D] bf16; O written [B,N,H,D] bf16. Block = 64 q-rows x 1 head.
// Q and K tiles staged by TDM (per-wave 16-row descriptors), K/V double-buffered.
__global__ void __launch_bounds__(128)
mla_attention_kernel(const uint16_t* __restrict__ Q, const uint16_t* __restrict__ K,
                     const uint16_t* __restrict__ V, const float* __restrict__ bias,
                     uint16_t* __restrict__ O) {
  __shared__ __attribute__((aligned(16))) uint16_t Qs[64][72];        // [q][d]
  __shared__ __attribute__((aligned(16))) uint16_t Ks[2][64][72];     // [kcol][d]
  __shared__ __attribute__((aligned(16))) uint16_t Vs[2][64][72];     // [d][k] transposed
  __shared__ __attribute__((aligned(16))) uint16_t Ps[4][16][72];     // wave-private P

  const int tid  = threadIdx.x;
  const int lane = tid & 31, wave = tid >> 5;
  const int hl   = lane >> 4, l16 = lane & 15;
  const int q0   = blockIdx.x * 64;
  const int h    = blockIdx.y;
  const int b    = blockIdx.z;
  const float scale = 0.125f;  // 1/sqrt(64)
  const uint32_t qsBase = (uint32_t)(uintptr_t)&Qs[0][0];
  const uint32_t ksBase = (uint32_t)(uintptr_t)&Ks[0][0][0];
  constexpr uint32_t kKVBufBytes = 64 * 72 * 2;

  auto stageKV = [&](int kt, int buf) {
    // K tile rows via TDM: row kc is kE elems apart (fixed head slice of [B,N,H,D])
    tdm_load_2d_bf16(ksBase + buf * kKVBufBytes + (uint32_t)wave * 16 * 144,
                     K + ((size_t)((b * kN + kt * 64 + wave * 16) * kH + h)) * kD,
                     64, 16, kE, kPad128to144);
    // V tile staged manually transposed: Vs[d][kc]
    for (int i = tid; i < 64 * 8; i += 128) {
      int r = i >> 3, c = i & 7;
      uint4 d = *(const uint4*)(V + ((size_t)((b * kN + kt * 64 + r) * kH + h)) * kD + c * 8);
      const uint16_t* ds = (const uint16_t*)&d;
#pragma unroll
      for (int e = 0; e < 8; ++e) Vs[buf][c * 8 + e][r] = ds[e];
    }
  };

  // Q tile (once, TDM) + first K/V tile
  tdm_load_2d_bf16(qsBase + (uint32_t)wave * 16 * 144,
                   Q + ((size_t)((b * kN + q0 + wave * 16) * kH + h)) * kD,
                   64, 16, kE, kPad128to144);
  stageKV(0, 0);
  __builtin_amdgcn_s_wait_tensorcnt(0);
  __syncthreads();

  const int kbA = hl ? 16 : 0;
  const int kbB = hl ? 16 : 0;
  FragB aq[2];  // Q A-fragments for the two 32-wide d-halves (resident all kernel)
#pragma unroll
  for (int s = 0; s < 2; ++s)
#pragma unroll
    for (int v = 0; v < 4; ++v) {
      aq[s].u[v]     = *(const uint32_t*)&Qs[wave * 16 + l16][s * 32 + kbA + 2 * v];
      aq[s].u[v + 4] = *(const uint32_t*)&Qs[wave * 16 + l16][s * 32 + 8 + kbA + 2 * v];
    }

  v8f oacc[4];
  float m[8], l[8];
#pragma unroll
  for (int dt = 0; dt < 4; ++dt) oacc[dt] = v8f_zero();
#pragma unroll
  for (int e = 0; e < 8; ++e) { m[e] = -3.0e38f; l[e] = 0.f; }

  for (int kt = 0; kt < kN / 64; ++kt) {
    const int cur = kt & 1;
    if (kt + 1 < kN / 64) stageKV(kt + 1, cur ^ 1);  // async next tile

    // S = Q * K^T * scale + bias : 4 column subtiles of 16
    float sfr[4][8];
#pragma unroll
    for (int ct = 0; ct < 4; ++ct) {
      FragB bk[2];
#pragma unroll
      for (int s = 0; s < 2; ++s)
#pragma unroll
        for (int v = 0; v < 8; ++v)
          bk[s].u[v] = *(const uint32_t*)&Ks[cur][ct * 16 + l16][s * 32 + kbB + 2 * v];
      v8f sv = v8f_zero();
      sv = __builtin_amdgcn_wmma_f32_16x16x32_bf16(false, aq[0].v, false, bk[0].v,
                                                   (short)0, sv, false, false);
      sv = __builtin_amdgcn_wmma_f32_16x16x32_bf16(false, aq[1].v, false, bk[1].v,
                                                   (short)0, sv, false, false);
#pragma unroll
      for (int e = 0; e < 8; ++e) {
        int row = q0 + wave * 16 + e + 8 * hl;
        int col = kt * 64 + ct * 16 + l16;
        sfr[ct][e] = sv[e] * scale + bias[(size_t)row * kN + col];
      }
    }

    // online softmax: row max over 64 cols (4 subtiles x 16 lanes in half-wave)
    float mnew[8];
#pragma unroll
    for (int e = 0; e < 8; ++e) {
      float t = fmaxf(fmaxf(sfr[0][e], sfr[1][e]), fmaxf(sfr[2][e], sfr[3][e]));
#pragma unroll
      for (int off = 1; off < 16; off <<= 1) t = fmaxf(t, __shfl_xor(t, off, 32));
      mnew[e] = fmaxf(t, m[e]);
    }
#pragma unroll
    for (int e = 0; e < 8; ++e) {
      float alpha = __expf(m[e] - mnew[e]);
      m[e] = mnew[e];
      l[e] *= alpha;
#pragma unroll
      for (int dt = 0; dt < 4; ++dt) oacc[dt][e] *= alpha;
    }
    float lsum[8];
#pragma unroll
    for (int e = 0; e < 8; ++e) lsum[e] = 0.f;
#pragma unroll
    for (int ct = 0; ct < 4; ++ct)
#pragma unroll
      for (int e = 0; e < 8; ++e) {
        float p = __expf(sfr[ct][e] - m[e]);
        lsum[e] += p;
        Ps[wave][e + 8 * hl][ct * 16 + l16] = f32_to_bf16(p);  // wave-private
      }
#pragma unroll
    for (int e = 0; e < 8; ++e) {
#pragma unroll
      for (int off = 1; off < 16; off <<= 1) lsum[e] += __shfl_xor(lsum[e], off, 32);
      l[e] += lsum[e];
    }

    // O += P * V
#pragma unroll
    for (int ks = 0; ks < 2; ++ks) {
      FragB ap;
#pragma unroll
      for (int v = 0; v < 4; ++v) {
        ap.u[v]     = *(const uint32_t*)&Ps[wave][l16][ks * 32 + kbA + 2 * v];
        ap.u[v + 4] = *(const uint32_t*)&Ps[wave][l16][ks * 32 + 8 + kbA + 2 * v];
      }
#pragma unroll
      for (int dt = 0; dt < 4; ++dt) {
        FragB bv;
#pragma unroll
        for (int v = 0; v < 8; ++v)
          bv.u[v] = *(const uint32_t*)&Vs[cur][dt * 16 + l16][ks * 32 + kbB + 2 * v];
        oacc[dt] = __builtin_amdgcn_wmma_f32_16x16x32_bf16(false, ap.v, false, bv.v,
                                                           (short)0, oacc[dt], false, false);
      }
    }

    __builtin_amdgcn_s_wait_tensorcnt(0);  // next K tile DMA complete
    __syncthreads();
  }

  // finalize: O /= l, write bf16 [B,N,H,D]
#pragma unroll
  for (int dt = 0; dt < 4; ++dt)
#pragma unroll
    for (int e = 0; e < 8; ++e) {
      int row = q0 + wave * 16 + e + 8 * hl;
      int col = dt * 16 + l16;
      float o = oacc[dt][e] / l[e];
      O[((size_t)((b * kN + row) * kH + h)) * kD + col] = f32_to_bf16(o);
    }
}

// ---------------- residual add + LayerNorm (block per row) ----------------
__global__ void __launch_bounds__(256)
add_ln_kernel(const float* __restrict__ X, const float* __restrict__ R,
              const float* __restrict__ g, const float* __restrict__ be,
              float* __restrict__ outf, uint16_t* __restrict__ outb) {
  const int row = blockIdx.x;
  const int tid = threadIdx.x;
  __shared__ float red[18];
  float v[4];
  float s = 0.f;
#pragma unroll
  for (int i = 0; i < 4; ++i) {
    int c = tid + i * 256;
    v[i] = X[(size_t)row * kE + c] + R[(size_t)row * kE + c];
    s += v[i];
  }
#pragma unroll
  for (int off = 1; off < 32; off <<= 1) s += __shfl_xor(s, off, 32);
  if ((tid & 31) == 0) red[tid >> 5] = s;
  __syncthreads();
  if (tid < 8) {
    float t = red[tid];
#pragma unroll
    for (int off = 1; off < 8; off <<= 1) t += __shfl_xor(t, off, 32);
    if (tid == 0) red[16] = t;
  }
  __syncthreads();
  float mean = red[16] * (1.f / kE);
  float s2 = 0.f;
#pragma unroll
  for (int i = 0; i < 4; ++i) { float d = v[i] - mean; s2 += d * d; }
#pragma unroll
  for (int off = 1; off < 32; off <<= 1) s2 += __shfl_xor(s2, off, 32);
  if ((tid & 31) == 0) red[8 + (tid >> 5)] = s2;
  __syncthreads();
  if (tid < 8) {
    float t = red[8 + tid];
#pragma unroll
    for (int off = 1; off < 8; off <<= 1) t += __shfl_xor(t, off, 32);
    if (tid == 0) red[17] = t;
  }
  __syncthreads();
  float inv = rsqrtf(red[17] * (1.f / kE) + 1e-5f);
#pragma unroll
  for (int i = 0; i < 4; ++i) {
    int c = tid + i * 256;
    float o = (v[i] - mean) * inv * g[c] + be[c];
    outf[(size_t)row * kE + c] = o;
    if (outb) outb[(size_t)row * kE + c] = f32_to_bf16(o);
  }
}

// ---------------- orchestration ----------------
extern "C" void kernel_launch(void* const* d_in, const int* in_sizes, int n_in,
                              void* d_out, int out_size, void* d_ws, size_t ws_size,
                              hipStream_t stream) {
  (void)in_sizes; (void)n_in; (void)out_size; (void)ws_size;
  const float* x     = (const float*)d_in[0];
  const float* attnb = (const float*)d_in[1];
  const float* Wq_w  = (const float*)d_in[2];
  const float* Wq_b  = (const float*)d_in[3];
  const float* WaK_w = (const float*)d_in[4];
  const float* WaK_b = (const float*)d_in[5];
  const float* WbK_w = (const float*)d_in[6];
  const float* WbK_b = (const float*)d_in[7];
  const float* WaV_w = (const float*)d_in[8];
  const float* WaV_b = (const float*)d_in[9];
  const float* WbV_w = (const float*)d_in[10];
  const float* WbV_b = (const float*)d_in[11];
  const float* Wo_w  = (const float*)d_in[12];
  const float* Wo_b  = (const float*)d_in[13];
  const float* f1_w  = (const float*)d_in[14];
  const float* f1_b  = (const float*)d_in[15];
  const float* f2_w  = (const float*)d_in[16];
  const float* f2_b  = (const float*)d_in[17];
  const float* ln1_g = (const float*)d_in[18];
  const float* ln1_b = (const float*)d_in[19];
  const float* ln2_g = (const float*)d_in[20];
  const float* ln2_b = (const float*)d_in[21];

  char* ws = (char*)d_ws;
  size_t off = 0;
  auto alloc16 = [&](size_t elems) {
    void* p = ws + off;
    off += (elems * 2 + 255) & ~(size_t)255;
    return (uint16_t*)p;
  };
  auto alloc32 = [&](size_t elems) {
    void* p = ws + off;
    off += (elems * 4 + 255) & ~(size_t)255;
    return (float*)p;
  };

  uint16_t* x16 = alloc16((size_t)kM * kE);
  uint16_t* wq  = alloc16((size_t)kE * kE);
  uint16_t* wak = alloc16((size_t)kE * kL);
  uint16_t* wbk = alloc16((size_t)kL * kE);
  uint16_t* wav = alloc16((size_t)kE * kL);
  uint16_t* wbv = alloc16((size_t)kL * kE);
  uint16_t* wo  = alloc16((size_t)kE * kE);
  uint16_t* wf1 = alloc16((size_t)kE * kF);
  uint16_t* wf2 = alloc16((size_t)kF * kE);
  uint16_t* Qb  = alloc16((size_t)kM * kE);
  uint16_t* Kc  = alloc16((size_t)kM * kL);
  uint16_t* Kb  = alloc16((size_t)kM * kE);
  uint16_t* Vc  = alloc16((size_t)kM * kL);
  uint16_t* Vb  = alloc16((size_t)kM * kE);
  uint16_t* Ob  = alloc16((size_t)kM * kE);
  float*    AO  = alloc32((size_t)kM * kE);
  float*    x1f = alloc32((size_t)kM * kE);
  uint16_t* x1b = alloc16((size_t)kM * kE);
  uint16_t* Hb  = alloc16((size_t)kM * kF);
  float*    Ff  = alloc32((size_t)kM * kE);

  auto cast = [&](const float* in, uint16_t* out, int n) {
    cast_f32_bf16_kernel<<<dim3((n + 1023) / 1024), dim3(256), 0, stream>>>(in, out, n);
  };
  cast(x, x16, kM * kE);
  cast(Wq_w, wq, kE * kE);
  cast(WaK_w, wak, kE * kL);
  cast(WbK_w, wbk, kL * kE);
  cast(WaV_w, wav, kE * kL);
  cast(WbV_w, wbv, kL * kE);
  cast(Wo_w, wo, kE * kE);
  cast(f1_w, wf1, kE * kF);
  cast(f2_w, wf2, kF * kE);

  dim3 blk(256);
  // Q = x @ Wq + b
  gemm_bf16_kernel<false><<<dim3(kE / 128, kM / 128), blk, 0, stream>>>(
      x16, wq, Wq_b, nullptr, Qb, kM, kE, kE);
  // K = (x @ WaK + b) @ WbK + b  (low-rank)
  gemm_bf16_kernel<false><<<dim3(kL / 128, kM / 128), blk, 0, stream>>>(
      x16, wak, WaK_b, nullptr, Kc, kM, kL, kE);
  gemm_bf16_kernel<false><<<dim3(kE / 128, kM / 128), blk, 0, stream>>>(
      Kc, wbk, WbK_b, nullptr, Kb, kM, kE, kL);
  // V = (x @ WaV + b) @ WbV + b
  gemm_bf16_kernel<false><<<dim3(kL / 128, kM / 128), blk, 0, stream>>>(
      x16, wav, WaV_b, nullptr, Vc, kM, kL, kE);
  gemm_bf16_kernel<false><<<dim3(kE / 128, kM / 128), blk, 0, stream>>>(
      Vc, wbv, WbV_b, nullptr, Vb, kM, kE, kL);
  // fused attention
  mla_attention_kernel<<<dim3(kN / 64, kH, kB), dim3(128), 0, stream>>>(Qb, Kb, Vb, attnb, Ob);
  // O @ Wo + b
  gemm_bf16_kernel<false><<<dim3(kE / 128, kM / 128), blk, 0, stream>>>(
      Ob, wo, Wo_b, AO, nullptr, kM, kE, kE);
  // x1 = LN(x + AO)
  add_ln_kernel<<<dim3(kM), dim3(256), 0, stream>>>(x, AO, ln1_g, ln1_b, x1f, x1b);
  // h = gelu(x1 @ W1 + b)
  gemm_bf16_kernel<true><<<dim3(kF / 128, kM / 128), blk, 0, stream>>>(
      x1b, wf1, f1_b, nullptr, Hb, kM, kF, kE);
  // ffn_out = h @ W2 + b
  gemm_bf16_kernel<false><<<dim3(kE / 128, kM / 128), blk, 0, stream>>>(
      Hb, wf2, f2_b, Ff, nullptr, kM, kE, kF);
  // out = LN(x1 + ffn_out)
  add_ln_kernel<<<dim3(kM), dim3(256), 0, stream>>>(x1f, Ff, ln2_g, ln2_b, (float*)d_out, nullptr);
}
